// TriMult_18502719111473
// MI455X (gfx1250) — matrix operand order repaired and checked by reading, not compiled, not run on hardware
//
#include <hip/hip_runtime.h>

// ---------------------------------------------------------------------------
// Triangle multiplication (outgoing) for MI455X / gfx1250.
// B=1, N=512, D=C=128.  All GEMM work on v_wmma_f32_16x16x32_bf16 (bf16 in,
// f32 accumulate).  left/right stored bf16 channel-major [c][i|j][k] so the
// einsum kernel gets contiguous-K fragments for BOTH operands; lt+rt = 128 MB
// -> L2 resident (192 MB).  tri stored channel-major [c][i][j] so every HBM
// line is written exactly once, coalesced.  tm_tri uses CDNA5 async
// global->LDS copies (ASYNCcnt) with double buffering when available.
// ---------------------------------------------------------------------------

static constexpr int N_  = 512;
static constexpr int D_  = 128;
static constexpr int C_  = 128;
static constexpr int NN_ = N_ * N_;

typedef __attribute__((ext_vector_type(16))) __bf16 v16bf;
typedef __attribute__((ext_vector_type(8)))  float  v8f;
typedef int v4i __attribute__((vector_size(16)));

#if __has_builtin(__builtin_amdgcn_global_load_async_to_lds_b128) && \
    __has_builtin(__builtin_amdgcn_s_wait_asynccnt)
#define TM_ASYNC 1
#else
#define TM_ASYNC 0
#endif

union FragU {
    uint4 q[2];
    v16bf v;
};

__device__ __forceinline__ unsigned short f2bf(float f) {
    union { float f; unsigned u; } v; v.f = f;
    unsigned r = v.u + 0x7FFFu + ((v.u >> 16) & 1u);   // round-to-nearest-even
    return (unsigned short)(r >> 16);
}
__device__ __forceinline__ float bf2f(unsigned short h) {
    union { unsigned u; float f; } v; v.u = ((unsigned)h) << 16; return v.f;
}
__device__ __forceinline__ float sigm(float x) {
    return 1.0f / (1.0f + __expf(-x));
}
__device__ __forceinline__ v8f zero8() {
    v8f z;
    #pragma unroll
    for (int i = 0; i < 8; ++i) z[i] = 0.0f;
    return z;
}
__device__ __forceinline__ v8f wmma_bf16(v16bf a, v16bf b, v8f c) {
    return __builtin_amdgcn_wmma_f32_16x16x32_bf16(
        false, a, false, b, (short)0, c, false, false);
}

// A fragment: two contiguous 16B K-chunks at +0 and +16 elements.
__device__ __forceinline__ v16bf ld_frag_split(const unsigned short* p0) {
    FragU u;
    u.q[0] = *(const uint4*)(p0);
    u.q[1] = *(const uint4*)(p0 + 16);
    return u.v;
}
// B fragment: one contiguous 32B K-chunk.
__device__ __forceinline__ v16bf ld_frag_contig(const unsigned short* p0) {
    FragU u;
    u.q[0] = *(const uint4*)(p0);
    u.q[1] = *(const uint4*)(p0 + 8);
    return u.v;
}
__device__ __forceinline__ v16bf ld_wfrag(const unsigned short* wpk,
                                          int mat, int kb, int half, int c) {
    return ld_frag_contig(wpk + (size_t)((((mat * 4 + kb) * 2 + half) * 128 + c)) * 16);
}

__device__ __forceinline__ uint4 pack8bf(const float* v) {
    uint4 r;
    r.x = (unsigned)f2bf(v[0]) | ((unsigned)f2bf(v[1]) << 16);
    r.y = (unsigned)f2bf(v[2]) | ((unsigned)f2bf(v[3]) << 16);
    r.z = (unsigned)f2bf(v[4]) | ((unsigned)f2bf(v[5]) << 16);
    r.w = (unsigned)f2bf(v[6]) | ((unsigned)f2bf(v[7]) << 16);
    return r;
}

#if TM_ASYNC
template <int OFF>
__device__ __forceinline__ void async_b128(const void* g, void* l) {
    __builtin_amdgcn_global_load_async_to_lds_b128(
        (__attribute__((address_space(1))) v4i*)(g),
        (__attribute__((address_space(3))) v4i*)(l), OFF, 0);
}
#define TM_WAIT_ASYNC(n) __builtin_amdgcn_s_wait_asynccnt(n)
#else
#define TM_WAIT_ASYNC(n)
#endif

// ---------------------------------------------------------------------------
// Kernel 0: repack 6 weight matrices (128x128 fp32) into bf16 B-fragment order:
// wpk[(((m*4+kb)*2+half)*128 + c)*16 + e] = W[kb*32+half*8+(e&7)+(e>>3)*16][c]
// ---------------------------------------------------------------------------
__global__ __launch_bounds__(256) void tm_pack(
    const float* __restrict__ Wl,  const float* __restrict__ Wr,
    const float* __restrict__ Wlg, const float* __restrict__ Wrg,
    const float* __restrict__ Wog, const float* __restrict__ Wo,
    unsigned short* __restrict__ wpk)
{
    int t = blockIdx.x * 256 + threadIdx.x;      // 98304 total
    int e    = t & 15;
    int c    = (t >> 4)  & 127;
    int half = (t >> 11) & 1;
    int kb   = (t >> 12) & 3;
    int m    = t >> 14;
    const float* Ws[6] = { Wl, Wr, Wlg, Wrg, Wog, Wo };
    int k = kb * 32 + half * 8 + (e & 7) + (e >> 3) * 16;
    wpk[t] = f2bf(Ws[m][k * 128 + c]);
}

// ---------------------------------------------------------------------------
// Kernel 1: LN(in) + 5 gated projections.  Block = fixed i, 16 consecutive k
// positions, so each lane's 8 results are K-contiguous -> single b128 store
// into lt[c][i][k] / rt[c][j][k].  Gate staged via LDS -> coalesced stores.
// ---------------------------------------------------------------------------
__global__ __launch_bounds__(256) void tm_proj(
    const float* __restrict__ x,
    const float* __restrict__ gin, const float* __restrict__ bin,
    const unsigned short* __restrict__ wpk,
    const float* __restrict__ bl,  const float* __restrict__ br,
    const float* __restrict__ blg, const float* __restrict__ brg,
    const float* __restrict__ bog,
    unsigned short* __restrict__ lt, unsigned short* __restrict__ rt,
    unsigned short* __restrict__ go)
{
    __shared__ unsigned short lnb[16 * 128];
    __shared__ unsigned short gob[16 * 128];
    __shared__ float part[16 * 16];
    __shared__ float redm[16];
    __shared__ float redv[16];

    const int t   = threadIdx.x;
    const int bx  = blockIdx.x;          // 512 * 32 blocks
    const int i   = bx & (N_ - 1);       // fixed second index
    const int k0  = (bx >> 9) * 16;      // 16 consecutive first-index rows
    const int row = t >> 4;              // position (k offset) within block
    const int q   = t & 15;

    // ---- LayerNorm over D=128 (deterministic tree reduction) ----
    float v[8];
    {
        const float4* xp =
            (const float4*)(x + ((size_t)(k0 + row) * N_ + i) * D_ + q * 8);
        float4 a = xp[0], b = xp[1];
        v[0]=a.x; v[1]=a.y; v[2]=a.z; v[3]=a.w;
        v[4]=b.x; v[5]=b.y; v[6]=b.z; v[7]=b.w;
    }
    float s = 0.f;
    #pragma unroll
    for (int e = 0; e < 8; ++e) s += v[e];
    part[row * 16 + q] = s;
    __syncthreads();
    if (q == 0) {
        float m = 0.f;
        for (int u = 0; u < 16; ++u) m += part[row * 16 + u];
        redm[row] = m * (1.0f / 128.0f);
    }
    __syncthreads();
    const float mean = redm[row];
    float vs = 0.f;
    #pragma unroll
    for (int e = 0; e < 8; ++e) { float d = v[e] - mean; vs += d * d; }
    part[row * 16 + q] = vs;
    __syncthreads();
    if (q == 0) {
        float m = 0.f;
        for (int u = 0; u < 16; ++u) m += part[row * 16 + u];
        redv[row] = rsqrtf(m * (1.0f / 128.0f) + 1e-5f);
    }
    __syncthreads();
    const float rstd = redv[row];
    #pragma unroll
    for (int e = 0; e < 8; ++e) {
        int c = q * 8 + e;
        lnb[row * 128 + c] = f2bf((v[e] - mean) * rstd * gin[c] + bin[c]);
    }
    __syncthreads();

    // ---- 8 waves x one 16-column tile; 5 GEMMs sharing A fragments ----
    const int w    = t >> 5;
    const int l    = t & 31;
    const int half = l >> 4;
    const int mr   = l & 15;
    const int c    = w * 16 + mr;

    v16bf a[4];
    #pragma unroll
    for (int kb = 0; kb < 4; ++kb)
        a[kb] = ld_frag_split(&lnb[mr * 128 + kb * 32 + half * 8]);

    v8f aL = zero8(), aR = zero8(), aLG = zero8(), aRG = zero8(), aOG = zero8();
    #pragma unroll
    for (int kb = 0; kb < 4; ++kb) {
        aL  = wmma_bf16(a[kb], ld_wfrag(wpk, 0, kb, half, c), aL);
        aR  = wmma_bf16(a[kb], ld_wfrag(wpk, 1, kb, half, c), aR);
        aLG = wmma_bf16(a[kb], ld_wfrag(wpk, 2, kb, half, c), aLG);
        aRG = wmma_bf16(a[kb], ld_wfrag(wpk, 3, kb, half, c), aRG);
        aOG = wmma_bf16(a[kb], ld_wfrag(wpk, 4, kb, half, c), aOG);
    }

    const float vbl  = bl[c],  vbr  = br[c];
    const float vblg = blg[c], vbrg = brg[c], vbog = bog[c];

    float lvv[8], rvv[8];
    #pragma unroll
    for (int r = 0; r < 8; ++r) {
        lvv[r] = sigm(aLG[r] + vblg) * (aL[r] + vbl);
        rvv[r] = sigm(aRG[r] + vbrg) * (aR[r] + vbr);
        gob[(r + 8 * half) * 128 + c] = f2bf(sigm(aOG[r] + vbog));
    }
    // lane's 8 values are K-contiguous: k = k0 + half*8 + r  -> one b128 each
    size_t base = (size_t)c * NN_ + (size_t)i * N_ + k0 + half * 8;
    *(uint4*)(lt + base) = pack8bf(lvv);
    *(uint4*)(rt + base) = pack8bf(rvv);

    __syncthreads();
    // coalesced gate store: go[p][d], p = (k0+row)*N + i
    *(uint4*)(go + ((size_t)(k0 + row) * N_ + i) * D_ + q * 8) =
        *(const uint4*)&gob[row * 128 + q * 8];
}

// ---------------------------------------------------------------------------
// Kernel 2: tri[c][i][j] = sum_k left[k,i,c] * right[k,j,c]
// 128x128 output tile per (i-tile, j-tile, channel) block.  K stepped by 32
// with double-buffered async global->LDS tile copies.  Wave w owns row-tile w
// (8 accumulators).  Output staged through LDS -> coalesced b128 row stores.
// ---------------------------------------------------------------------------
__global__ __launch_bounds__(256) void tm_tri(
    const unsigned short* __restrict__ lt,
    const unsigned short* __restrict__ rt,
    float* __restrict__ tri)
{
    __shared__ __align__(16) unsigned char smem[32768];
    unsigned short* Lsb[2] = { (unsigned short*)smem,
                               (unsigned short*)(smem + 8192) };
    unsigned short* Rsb[2] = { (unsigned short*)(smem + 16384),
                               (unsigned short*)(smem + 24576) };

    const int t  = threadIdx.x;
    const int c  = blockIdx.z;
    const int i0 = blockIdx.x * 128;
    const int j0 = blockIdx.y * 128;
    const unsigned short* ltc = lt + (size_t)c * NN_;
    const unsigned short* rtc = rt + (size_t)c * NN_;

    const int w    = t >> 5;          // wave -> row tile
    const int l    = t & 31;
    const int half = l >> 4;
    const int mr   = l & 15;

    const int lrow = t >> 1;          // 0..127: LDS tile row this thread loads
    const int lkh  = (t & 1) * 16;    // K half chunk

    v8f acc[8];
    #pragma unroll
    for (int n = 0; n < 8; ++n) acc[n] = zero8();

    // issue tile loads for k-step kk into buffer b
    auto load_tiles = [&](int kk, int b) {
        const unsigned short* gL = ltc + (size_t)(i0 + lrow) * N_ + kk * 32 + lkh;
        const unsigned short* gR = rtc + (size_t)(j0 + lrow) * N_ + kk * 32 + lkh;
        unsigned short* lL = Lsb[b] + lrow * 32 + lkh;
        unsigned short* lR = Rsb[b] + lrow * 32 + lkh;
#if TM_ASYNC
        async_b128<0>(gL, lL);  async_b128<16>(gL, lL);
        async_b128<0>(gR, lR);  async_b128<16>(gR, lR);
#else
        *(uint4*)lL       = *(const uint4*)gL;
        *(uint4*)(lL + 8) = *(const uint4*)(gL + 8);
        *(uint4*)lR       = *(const uint4*)gR;
        *(uint4*)(lR + 8) = *(const uint4*)(gR + 8);
#endif
    };

    load_tiles(0, 0);
    for (int kk = 0; kk < 16; ++kk) {            // K = 512, step 32
        const int cur = kk & 1;
        if (kk < 15) {
            load_tiles(kk + 1, cur ^ 1);
            TM_WAIT_ASYNC(4);                    // 4 newest (next buf) pending
        } else {
            TM_WAIT_ASYNC(0);
        }
        __syncthreads();

        v16bf a = ld_frag_split(&Lsb[cur][(w * 16 + mr) * 32 + half * 8]);
        #pragma unroll
        for (int n = 0; n < 8; ++n) {
            v16bf b = ld_frag_split(&Rsb[cur][(n * 16 + mr) * 32 + half * 8]);
            acc[n] = wmma_bf16(a, b, acc[n]);
        }
        __syncthreads();
    }

    // ---- stage through LDS, store coalesced rows of tri[c][i][j] ----
    float* stage = (float*)smem;                 // 32 KB = 4 row tiles
    const size_t cb = (size_t)c * NN_;
    #pragma unroll
    for (int pass = 0; pass < 2; ++pass) {
        __syncthreads();
        if ((w >> 2) == pass) {
            int tloc = w & 3;
            #pragma unroll
            for (int n = 0; n < 8; ++n)
                #pragma unroll
                for (int r = 0; r < 8; ++r)
                    stage[tloc * 2048 + (r + 8 * half) * 128 + n * 16 + mr] =
                        acc[n][r];
        }
        __syncthreads();
        const int rrow = t >> 2;                 // 0..63
        const int qq   = t & 3;
        const float* srow = stage + rrow * 128 + qq * 32;
        float* gdst = tri + cb + (size_t)(i0 + pass * 64 + rrow) * N_ + j0 + qq * 32;
        #pragma unroll
        for (int u = 0; u < 8; ++u)
            *(float4*)(gdst + u * 4) = *(const float4*)(srow + u * 4);
    }
}

// ---------------------------------------------------------------------------
// Kernel 3: out = gate * (LN_C(tri) @ Wo + bo), 16 positions per block.
// tri is [c][p]: transpose 128c x 16p slab through padded LDS, LN, WMMA,
// then stage result in LDS for coalesced gate-multiply + float4 stores.
// ---------------------------------------------------------------------------
__global__ __launch_bounds__(256) void tm_out(
    const float* __restrict__ tri,
    const float* __restrict__ gout, const float* __restrict__ bout,
    const unsigned short* __restrict__ wpk,
    const float* __restrict__ bo,
    const unsigned short* __restrict__ go,
    float* __restrict__ out)
{
    constexpr int PITCH = 17;                    // bank-conflict padding
    __shared__ float trS[128 * PITCH];
    __shared__ unsigned short lnb[16 * 128];
    __shared__ float outs[16 * 128];
    __shared__ float part[16 * 16];
    __shared__ float redm[16];
    __shared__ float redv[16];

    const int t   = threadIdx.x;
    const int p0  = blockIdx.x * 16;
    const int row = t >> 4;
    const int q   = t & 15;

    // ---- transpose-load: tri[c][p0..p0+15] -> trS[c][row] ----
    {
        const int cidx = t >> 1;                 // 0..127
        const int qq   = t & 1;
        const float* gsrc = tri + (size_t)cidx * NN_ + p0 + qq * 8;
        float4 a4 = ((const float4*)gsrc)[0];
        float4 b4 = ((const float4*)gsrc)[1];
        float* d = trS + cidx * PITCH + qq * 8;
        d[0]=a4.x; d[1]=a4.y; d[2]=a4.z; d[3]=a4.w;
        d[4]=b4.x; d[5]=b4.y; d[6]=b4.z; d[7]=b4.w;
    }
    __syncthreads();

    float v[8];
    #pragma unroll
    for (int e = 0; e < 8; ++e) v[e] = trS[(q * 8 + e) * PITCH + row];

    float s = 0.f;
    #pragma unroll
    for (int e = 0; e < 8; ++e) s += v[e];
    part[row * 16 + q] = s;
    __syncthreads();
    if (q == 0) {
        float m = 0.f;
        for (int u = 0; u < 16; ++u) m += part[row * 16 + u];
        redm[row] = m * (1.0f / 128.0f);
    }
    __syncthreads();
    const float mean = redm[row];
    float vs = 0.f;
    #pragma unroll
    for (int e = 0; e < 8; ++e) { float d = v[e] - mean; vs += d * d; }
    part[row * 16 + q] = vs;
    __syncthreads();
    if (q == 0) {
        float m = 0.f;
        for (int u = 0; u < 16; ++u) m += part[row * 16 + u];
        redv[row] = rsqrtf(m * (1.0f / 128.0f) + 1e-5f);
    }
    __syncthreads();
    const float rstd = redv[row];
    #pragma unroll
    for (int e = 0; e < 8; ++e) {
        int c = q * 8 + e;
        lnb[row * 128 + c] = f2bf((v[e] - mean) * rstd * gout[c] + bout[c]);
    }
    __syncthreads();

    const int w    = t >> 5;
    const int l    = t & 31;
    const int half = l >> 4;
    const int mr   = l & 15;
    const int d    = w * 16 + mr;

    v16bf a[4];
    #pragma unroll
    for (int kb = 0; kb < 4; ++kb)
        a[kb] = ld_frag_split(&lnb[mr * 128 + kb * 32 + half * 8]);

    v8f acc = zero8();
    #pragma unroll
    for (int kb = 0; kb < 4; ++kb)
        acc = wmma_bf16(a[kb], ld_wfrag(wpk, 5, kb, half, d), acc);

    const float vbo = bo[d];
    #pragma unroll
    for (int r = 0; r < 8; ++r)
        outs[(r + 8 * half) * 128 + d] = acc[r] + vbo;
    __syncthreads();

    // ---- gate multiply + coalesced store ----
    const size_t p = (size_t)(p0 + row);
    uint4 gq = *(const uint4*)&go[p * D_ + q * 8];
    const unsigned short* gs = (const unsigned short*)&gq;
    float o[8];
    #pragma unroll
    for (int e = 0; e < 8; ++e)
        o[e] = outs[row * 128 + q * 8 + e] * bf2f(gs[e]);
    float4 r0 = { o[0], o[1], o[2], o[3] };
    float4 r1 = { o[4], o[5], o[6], o[7] };
    float4* gd = (float4*)(out + p * D_ + q * 8);
    gd[0] = r0;
    gd[1] = r1;
}

// ---------------------------------------------------------------------------
extern "C" void kernel_launch(void* const* d_in, const int* in_sizes, int n_in,
                              void* d_out, int out_size, void* d_ws, size_t ws_size,
                              hipStream_t stream)
{
    const float* x    = (const float*)d_in[0];
    const float* ging = (const float*)d_in[1];
    const float* binb = (const float*)d_in[2];
    const float* gout = (const float*)d_in[3];
    const float* bout = (const float*)d_in[4];
    const float* Wl   = (const float*)d_in[5];
    const float* bl   = (const float*)d_in[6];
    const float* Wr   = (const float*)d_in[7];
    const float* br   = (const float*)d_in[8];
    const float* Wo   = (const float*)d_in[9];
    const float* bo   = (const float*)d_in[10];
    const float* Wlg  = (const float*)d_in[11];
    const float* blg  = (const float*)d_in[12];
    const float* Wrg  = (const float*)d_in[13];
    const float* brg  = (const float*)d_in[14];
    const float* Wog  = (const float*)d_in[15];
    const float* bog  = (const float*)d_in[16];

    char*  ws  = (char*)d_ws;
    size_t off = 0;
    auto take = [&](size_t bytes) {
        size_t o = off;
        off += (bytes + 255) & ~(size_t)255;
        return o;
    };
    unsigned short* wpk = (unsigned short*)(ws + take((size_t)6 * 128 * 128 * 2));
    unsigned short* lt  = (unsigned short*)(ws + take((size_t)C_ * NN_ * 2));
    unsigned short* rt  = (unsigned short*)(ws + take((size_t)C_ * NN_ * 2));
    unsigned short* go  = (unsigned short*)(ws + take((size_t)NN_ * D_ * 2));
    float*          tri = (float*)(ws + take((size_t)NN_ * C_ * 4));
    (void)ws_size; (void)in_sizes; (void)n_in; (void)out_size;

    tm_pack<<<384, 256, 0, stream>>>(Wl, Wr, Wlg, Wrg, Wog, Wo, wpk);
    tm_proj<<<NN_ / 16, 256, 0, stream>>>(x, ging, binb, wpk,
                                          bl, br, blg, brg, bog, lt, rt, go);
    dim3 gB(N_ / 128, N_ / 128, C_);
    tm_tri<<<gB, 256, 0, stream>>>(lt, rt, tri);
    tm_out<<<NN_ / 16, 256, 0, stream>>>(tri, gout, bout, wpk, bo, go,
                                         (float*)d_out);
}
